// PairwisePredictor_29978871726277
// MI455X (gfx1250) — matrix operand order repaired and checked by reading, not compiled, not run on hardware
//
#include <hip/hip_runtime.h>
#include <cstddef>
#include <cstdint>

// ---------------------------------------------------------------------------
// PairwisePredictor on MI455X (gfx1250, wave32, WMMA f16 16x16x32 + TDM)
// Fused pipeline: never materializes (384,384,512) pairwise / ax tensors.
// ---------------------------------------------------------------------------

typedef _Float16 half16 __attribute__((ext_vector_type(16)));
typedef _Float16 half8  __attribute__((ext_vector_type(8)));
typedef float    floatx8 __attribute__((ext_vector_type(8)));
typedef float    f32x4  __attribute__((ext_vector_type(4)));
typedef unsigned int u32x4 __attribute__((ext_vector_type(4)));
typedef unsigned int u32x8 __attribute__((ext_vector_type(8)));

#define NOBJ 128
#define NREL 384
#define HID  512
#define POOL 4096
#define NRELCLS 51
#define NOBJCLS 151

static __device__ __forceinline__ floatx8 wmma_f16(half16 a, half16 b, floatx8 c) {
  return __builtin_amdgcn_wmma_f32_16x16x32_f16(false, a, false, b, (short)0, c,
                                                false, false);
}

// f16 A-frag in this code's contiguous k order (16x32 row-major, ld halves, ld % 8 == 0):
// lane L: M=L%16, kb=(L/16)*16 ; halves = K[kb..kb+15]
// -> one contiguous 32B run -> 2x ds_load_b128
static __device__ __forceinline__ half16 load_a_frag(const _Float16* p, int ld) {
  int lane = threadIdx.x & 31;
  int m = lane & 15;
  int kb = (lane >> 4) << 4;
  half8 lo = *(const half8*)(p + m * ld + kb);
  half8 hi = *(const half8*)(p + m * ld + kb + 8);
  return __builtin_shufflevector(lo, hi, 0, 1, 2, 3, 4, 5, 6, 7,
                                 8, 9, 10, 11, 12, 13, 14, 15);
}

// f16 B-frag (32x16, contiguous k order as in A) from N-major (column-major) LDS: value(k,n) at
// p[n*ldk + k]. lane L: N=L%16, kb=(L/16)*16 ; halves = K[kb..kb+15]
// -> one contiguous 32B run -> 2x ds_load_b128
static __device__ __forceinline__ half16 load_b_frag_cm(const _Float16* p, int ldk) {
  int lane = threadIdx.x & 31;
  int n = lane & 15;
  int kb = (lane >> 4) << 4;
  const half8* q = (const half8*)(p + n * ldk + kb);
  half8 lo = q[0];
  half8 hi = q[1];
  return __builtin_shufflevector(lo, hi, 0, 1, 2, 3, 4, 5, 6, 7,
                                 8, 9, 10, 11, 12, 13, 14, 15);
}

// C/D frag mapping: element r -> (M = r + 8*(lane/16), N = lane%16)

// ---------------------------------------------------------------------------
// Tensor Data Mover: 3D f16 tile (32 x 16 x 16 elements) -> LDS, with LDS
// padding of 64B after every 64B line (interleaves two tiles into 128B rows).
// All descriptor fields are wave-uniform -> SGPR groups for VIMAGE encoding.
// ---------------------------------------------------------------------------
static __device__ __forceinline__ void tdm_load_tile_f16(
    unsigned int lds_addr, unsigned long long gaddr,
    unsigned int d0_stride, unsigned int d1_stride) {
  u32x4 g0;
  g0[0] = 1u;                                   // count=1 (valid), user mode
  g0[1] = lds_addr;                             // lds_addr (bytes)
  g0[2] = (unsigned int)gaddr;                  // global_addr[31:0]
  g0[3] = (unsigned int)(gaddr >> 32) | (2u << 30);  // global_addr[56:32] | type=2
  u32x8 g1;
  g1[0] = (1u << 16)        // data_size = 1 -> 2 bytes
        | (1u << 20)        // pad_enable
        | (3u << 22)        // pad_interval: 2^(3+1)=16 DWORDs (64B)
        | (15u << 25);      // pad_amount: 16 DWORDs (64B)
  g1[1] = (32u << 16);      // tensor_dim0 = 32 (bits 63:48)
  g1[2] = (16u << 16);      // tensor_dim1 = 16 (bits 95:80)
  g1[3] = (32u << 16);      // tile_dim0 = 32 (bits 127:112)
  g1[4] = 16u | (16u << 16);// tile_dim1 = 16, tile_dim2 = 16
  g1[5] = d0_stride;        // tensor_dim0_stride[31:0]
  g1[6] = (d1_stride & 0xFFFFu) << 16;  // d0_stride[47:32]=0 | d1_stride[15:0]
  g1[7] = d1_stride >> 16;  // tensor_dim1_stride[47:16]
  u32x4 g2;
  g2[0] = 16u;              // tensor_dim2 = 16
  g2[1] = 0u; g2[2] = 0u; g2[3] = 0u;
  u32x4 g3; g3[0] = 0u; g3[1] = 0u; g3[2] = 0u; g3[3] = 0u;
  asm volatile("tensor_load_to_lds %0, %1, %2, %3"
               :: "s"(g0), "s"(g1), "s"(g2), "s"(g3)
               : "memory");
}

// ---------------------------------------------------------------------------
// Generic WMMA GEMM: C[M,N] (+)= act(A[M,K] * B[K,N] + bias)
// block = 128 threads (4 waves), 32x32 tile. Requires M%32==0, K%32==0.
// ---------------------------------------------------------------------------
__global__ void pp_gemm_kernel(const float* __restrict__ A, int lda,
                               const float* __restrict__ B, int ldb,
                               const float* __restrict__ bias,
                               float* __restrict__ C, int ldc,
                               int M, int N, int K, int do_relu, int do_acc) {
  __shared__ __align__(16) _Float16 As[32 * 32];   // row-major [m][k]
  __shared__ __align__(16) _Float16 BsT[32 * 32];  // N-major  [n][k]
  const int tid = threadIdx.x;
  const int lane = tid & 31, wave = tid >> 5;
  const int n0 = blockIdx.x * 32, m0 = blockIdx.y * 32;
  const int wm = (wave >> 1) << 4, wn = (wave & 1) << 4;
  floatx8 acc = {};
  // staging roles
  const int ar = tid >> 2, ak8 = (tid & 3) << 3;          // A: row, k-chunk
  const int bn = tid & 31, bk8 = (tid >> 5) << 3;         // B: col, k-chunk
  const int gn = n0 + bn;
  const bool bvalid = (gn < N);                           // hoisted guard
  for (int kc = 0; kc < K; kc += 32) {
    if (kc + 32 < K)
      __builtin_prefetch((const void*)(B + (size_t)(kc + 32) * ldb + n0), 0, 1);
    {  // A tile: 2x global_load_b128 -> half8 -> 1x ds_store_b128
      const f32x4* src = (const f32x4*)(A + (size_t)(m0 + ar) * lda + kc + ak8);
      f32x4 u = src[0], v = src[1];
      half8 h;
      h[0] = (_Float16)u[0]; h[1] = (_Float16)u[1];
      h[2] = (_Float16)u[2]; h[3] = (_Float16)u[3];
      h[4] = (_Float16)v[0]; h[5] = (_Float16)v[1];
      h[6] = (_Float16)v[2]; h[7] = (_Float16)v[3];
      *(half8*)(As + ar * 32 + ak8) = h;
    }
    {  // B tile transposed: 8 lane-coalesced b32 loads (one clause) -> b128
      half8 h = {};
      if (bvalid) {
#pragma unroll
        for (int e = 0; e < 8; ++e) {
          float v = B[(size_t)(kc + bk8 + e) * ldb + gn];
          h[e] = (_Float16)v;
        }
      }
      *(half8*)(BsT + bn * 32 + bk8) = h;
    }
    __syncthreads();
    half16 a = load_a_frag(As + wm * 32, 32);
    half16 b = load_b_frag_cm(BsT + wn * 32, 32);
    acc = wmma_f16(a, b, acc);
    __syncthreads();
  }
  int n = n0 + wn + (lane & 15);
  if (n < N) {
    float bv = bias ? bias[n] : 0.f;
#pragma unroll
    for (int r = 0; r < 8; ++r) {
      int m = m0 + wm + r + ((lane >> 4) << 3);
      float v = acc[r] + bv;
      if (do_relu) v = fmaxf(v, 0.f);
      size_t off = (size_t)m * ldc + n;
      if (do_acc) C[off] += v; else C[off] = v;
    }
  }
}

// ---------------------------------------------------------------------------
// bn1d over rows
// ---------------------------------------------------------------------------
__global__ void pp_bn1d_kernel(const float* __restrict__ X, float* __restrict__ Y,
                               int R, int C, const float* __restrict__ g,
                               const float* __restrict__ be) {
  int c = blockIdx.x * blockDim.x + threadIdx.x;
  if (c >= C) return;
  float s = 0.f, s2 = 0.f;
  for (int r = 0; r < R; ++r) { float v = X[(size_t)r * C + c]; s += v; s2 += v * v; }
  float m = s / (float)R;
  float var = s2 / (float)R - m * m;
  float rs = rsqrtf(var + 1e-5f) * g[c];
  float b = be[c];
  for (int r = 0; r < R; ++r) Y[(size_t)r * C + c] = (X[(size_t)r * C + c] - m) * rs + b;
}

// Y = bn(X1) + bn(X2) + P   (ctx_pairwise_rois)
__global__ void pp_bn_add_kernel(const float* __restrict__ X1, const float* __restrict__ X2,
                                 const float* __restrict__ P,
                                 const float* __restrict__ g1, const float* __restrict__ be1,
                                 const float* __restrict__ g2, const float* __restrict__ be2,
                                 float* __restrict__ Y) {
  int c = blockIdx.x * blockDim.x + threadIdx.x;
  if (c >= HID) return;
  float s1 = 0.f, q1 = 0.f, s2 = 0.f, q2 = 0.f;
  for (int r = 0; r < NREL; ++r) {
    float a = X1[(size_t)r * HID + c]; s1 += a; q1 += a * a;
    float b = X2[(size_t)r * HID + c]; s2 += b; q2 += b * b;
  }
  const float inv = 1.f / (float)NREL;
  float m1 = s1 * inv, v1 = q1 * inv - m1 * m1;
  float m2 = s2 * inv, v2 = q2 * inv - m2 * m2;
  float r1 = rsqrtf(v1 + 1e-5f) * g1[c];
  float r2 = rsqrtf(v2 + 1e-5f) * g2[c];
  float b1 = be1[c], b2 = be2[c];
  for (int r = 0; r < NREL; ++r) {
    size_t o = (size_t)r * HID + c;
    Y[o] = (X1[o] - m1) * r1 + b1 + (X2[o] - m2) * r2 + b2 + P[o];
  }
}

// ---------------------------------------------------------------------------
// Gather s/o rows (vectorized 8 cols per thread)
// ---------------------------------------------------------------------------
__global__ void pp_gather_kernel(const float* __restrict__ obj_ctx,
                                 const int* __restrict__ pairs,
                                 float* __restrict__ concat_so,
                                 _Float16* __restrict__ sh, _Float16* __restrict__ oh) {
  int chunk = blockIdx.x * 256 + threadIdx.x;          // 24576 chunks of 8
  if (chunk >= NREL * HID / 8) return;
  int r = chunk >> 6, c8 = (chunk & 63) << 3;
  int hidx = pairs[r * 2 + 0], tidx = pairs[r * 2 + 1];
  const f32x4* sp = (const f32x4*)(obj_ctx + (size_t)hidx * HID + c8);
  const f32x4* op = (const f32x4*)(obj_ctx + (size_t)tidx * HID + c8);
  f32x4 s0 = sp[0], s1 = sp[1], o0 = op[0], o1 = op[1];
  *(f32x4*)(concat_so + (size_t)r * 1024 + c8) = s0;
  *(f32x4*)(concat_so + (size_t)r * 1024 + c8 + 4) = s1;
  *(f32x4*)(concat_so + (size_t)r * 1024 + HID + c8) = o0;
  *(f32x4*)(concat_so + (size_t)r * 1024 + HID + c8 + 4) = o1;
  half8 hs, ho;
  hs[0]=(_Float16)s0[0]; hs[1]=(_Float16)s0[1]; hs[2]=(_Float16)s0[2]; hs[3]=(_Float16)s0[3];
  hs[4]=(_Float16)s1[0]; hs[5]=(_Float16)s1[1]; hs[6]=(_Float16)s1[2]; hs[7]=(_Float16)s1[3];
  ho[0]=(_Float16)o0[0]; ho[1]=(_Float16)o0[1]; ho[2]=(_Float16)o0[2]; ho[3]=(_Float16)o0[3];
  ho[4]=(_Float16)o1[0]; ho[5]=(_Float16)o1[1]; ho[6]=(_Float16)o1[2]; ho[7]=(_Float16)o1[3];
  *(half8*)(sh + (size_t)r * HID + c8) = hs;
  *(half8*)(oh + (size_t)r * HID + c8) = ho;
}

// ---------------------------------------------------------------------------
// Fused axial self-attention branch. One workgroup = one batch element b.
// Needs ~110KB dynamic LDS (CDNA5 320KB WGP LDS).
// ---------------------------------------------------------------------------
__global__ void pp_axial_attn_kernel(const _Float16* __restrict__ rows_h,
                                     const _Float16* __restrict__ gate_h,
                                     const float* __restrict__ Wq,   // HIDx32
                                     const float* __restrict__ Wkv,  // HIDx64
                                     _Float16* __restrict__ outp) {  // [384][384][32]
  extern __shared__ __align__(16) char smem[];
  _Float16* xh  = (_Float16*)smem;        // [384][32] row-major (A)
  _Float16* whT = xh + NREL * 32;         // [96][32]  N-major  (B)
  _Float16* qh  = whT + 96 * 32;          // [384][32] row-major (A)
  _Float16* kB  = qh + NREL * 32;         // [384][32] N-major: (k=e,n=i') (B)
  _Float16* vT  = kB + NREL * 32;         // [32][384] N-major: (k=i',n=e) (B)
  _Float16* pst = vT + 32 * NREL;         // 8 waves * [16][32] staging (A)

  const int b = blockIdx.x;
  const int tid = threadIdx.x, lane = tid & 31, wave = tid >> 5;
  const _Float16* grow = gate_h + (size_t)b * HID;

  floatx8 acc[3][6];
#pragma unroll
  for (int i = 0; i < 3; ++i)
#pragma unroll
    for (int j = 0; j < 6; ++j) { floatx8 z = {}; acc[i][j] = z; }

  // ---- Phase 1: QKV projection (M=384, N=96, K=HID in chunks of 32) ----
  for (int kc = 0; kc < HID; kc += 32) {
    for (int x = tid; x < NREL * 4; x += 256) {   // 8-half chunks
      int i = x >> 2, k8 = (x & 3) << 3;
      half8 a = *(const half8*)(rows_h + (size_t)i * HID + kc + k8);
      half8 g = *(const half8*)(grow + kc + k8);
      *(half8*)(xh + i * 32 + k8) = a * g;        // v_pk_mul_f16
    }
    if (tid < 96) {                               // whT[n][k], coalesced over n
      int n = tid;
      if (n < 32) {                               // uniform-per-thread branch
#pragma unroll
        for (int g8 = 0; g8 < 4; ++g8) {
          half8 h;
#pragma unroll
          for (int e = 0; e < 8; ++e) h[e] = (_Float16)Wq[(kc + g8 * 8 + e) * 32 + n];
          *(half8*)(whT + n * 32 + g8 * 8) = h;
        }
      } else {
        int n2 = n - 32;
#pragma unroll
        for (int g8 = 0; g8 < 4; ++g8) {
          half8 h;
#pragma unroll
          for (int e = 0; e < 8; ++e) h[e] = (_Float16)Wkv[(kc + g8 * 8 + e) * 64 + n2];
          *(half8*)(whT + n * 32 + g8 * 8) = h;
        }
      }
    }
    __syncthreads();
#pragma unroll
    for (int mt = 0; mt < 3; ++mt) {
      half16 a = load_a_frag(xh + (wave * 3 + mt) * 16 * 32, 32);
#pragma unroll
      for (int nt = 0; nt < 6; ++nt) {
        half16 bf = load_b_frag_cm(whT + nt * 16 * 32, 32);
        acc[mt][nt] = wmma_f16(a, bf, acc[mt][nt]);
      }
    }
    __syncthreads();
  }
  // scatter q/k/v into WMMA-operand layouts
#pragma unroll
  for (int mt = 0; mt < 3; ++mt) {
    int mbase = (wave * 3 + mt) * 16;
#pragma unroll
    for (int nt = 0; nt < 6; ++nt) {
#pragma unroll
      for (int r = 0; r < 8; ++r) {
        int m = mbase + r + ((lane >> 4) << 3);
        int n = nt * 16 + (lane & 15);
        _Float16 v = (_Float16)acc[mt][nt][r];
        if (n < 32)       qh[m * 32 + n] = v;
        else if (n < 64)  kB[m * 32 + (n - 32)] = v;  // row i' holds k-column
        else              vT[(n - 64) * NREL + m] = v;
      }
    }
  }
  __syncthreads();

  // ---- Phase 2: per wave 3 query tiles; two-pass softmax (scores O(1)) ----
  const float scale = 0.17677669529663687f;  // 1/sqrt(32)
  _Float16* mypst = pst + wave * 16 * 32;
  for (int t = 0; t < 3; ++t) {
    int row0 = (wave * 3 + t) * 16;
    half16 aq = load_a_frag(qh + row0 * 32, 32);
    float rsum[8];
#pragma unroll
    for (int r = 0; r < 8; ++r) rsum[r] = 0.f;
    for (int jt = 0; jt < 24; ++jt) {
      half16 bk = load_b_frag_cm(kB + jt * 16 * 32, 32);
      floatx8 z = {};
      floatx8 d = wmma_f16(aq, bk, z);
#pragma unroll
      for (int r = 0; r < 8; ++r) {
        float e = __expf(d[r] * scale);
        e += __shfl_xor(e, 1); e += __shfl_xor(e, 2);
        e += __shfl_xor(e, 4); e += __shfl_xor(e, 8);
        rsum[r] += e;
      }
    }
    floatx8 oacc0 = {}, oacc1 = {};
    for (int jc = 0; jc < 12; ++jc) {
      __syncthreads();  // uniform across waves; orders pst reuse
#pragma unroll
      for (int h2 = 0; h2 < 2; ++h2) {
        int jt = jc * 2 + h2;
        half16 bk = load_b_frag_cm(kB + jt * 16 * 32, 32);
        floatx8 z = {};
        floatx8 d = wmma_f16(aq, bk, z);
#pragma unroll
        for (int r = 0; r < 8; ++r) {
          float p = __expf(d[r] * scale) / rsum[r];
          int mr = r + ((lane >> 4) << 3);
          mypst[mr * 32 + h2 * 16 + (lane & 15)] = (_Float16)p;
        }
      }
      __syncthreads();
      half16 ap  = load_a_frag(mypst, 32);
      half16 bv0 = load_b_frag_cm(vT + jc * 32, NREL);
      half16 bv1 = load_b_frag_cm(vT + 16 * NREL + jc * 32, NREL);
      oacc0 = wmma_f16(ap, bv0, oacc0);
      oacc1 = wmma_f16(ap, bv1, oacc1);
    }
#pragma unroll
    for (int r = 0; r < 8; ++r) {
      int m = row0 + r + ((lane >> 4) << 3);
      size_t base = ((size_t)b * NREL + m) * 32;
      outp[base + (lane & 15)]      = (_Float16)oacc0[r];
      outp[base + 16 + (lane & 15)] = (_Float16)oacc1[r];
    }
  }
}

// ---------------------------------------------------------------------------
// Combine: ax[i,j,c] = relu([out1[j,i],out2[i,j]] @ [Wo1;Wo2] + bo1+bo2)
// A-tiles staged by the Tensor Data Mover (2 interleaved 3D tiles per iter).
// One block owns 16 j's x all 384 i's; pairs ordered p = jl*16 + il (TDM
// Z-outer order) so each lane's 8 C-rows share one jl -> 1 ds_add_f32.
// ---------------------------------------------------------------------------
__global__ void pp_combine_kernel(const _Float16* __restrict__ out1,
                                  const _Float16* __restrict__ out2,
                                  const float* __restrict__ Wo1,  // 32x512
                                  const float* __restrict__ Wo2,  // 32x512
                                  const float* __restrict__ bo1,
                                  const float* __restrict__ bo2,
                                  float* __restrict__ Sglob,      // [384][512]
                                  float* __restrict__ chanSum,
                                  float* __restrict__ chanSumSq) {
  extern __shared__ __align__(16) char smem[];
  _Float16* BwT = (_Float16*)smem;           // [512 n][64 k] N-major (B)
  _Float16* At  = BwT + 64 * HID;            // [256 p][64 k] row-major (A)
  float* S  = (float*)(At + 256 * 64);       // [16 jl][512]
  float* s1 = S + 16 * HID;                  // [512]
  float* s2 = s1 + HID;                      // [512]
  float* bv = s2 + HID;                      // [512]
  const int tid = threadIdx.x, lane = tid & 31, wave = tid >> 5;
  const int jb = blockIdx.x;                 // 0..23

  for (int n = tid; n < HID; n += 256) {     // BwT[n][k], coalesced over n
#pragma unroll
    for (int g8 = 0; g8 < 8; ++g8) {
      half8 h;
#pragma unroll
      for (int e = 0; e < 8; ++e) {
        int k = g8 * 8 + e;  // compile-time constant select Wo1/Wo2
        float v = (k < 32) ? Wo1[k * HID + n] : Wo2[(k - 32) * HID + n];
        h[e] = (_Float16)v;
      }
      *(half8*)(BwT + n * 64 + g8 * 8) = h;
    }
  }
  for (int c = tid; c < HID; c += 256) { bv[c] = bo1[c] + bo2[c]; s1[c] = 0.f; s2[c] = 0.f; }
  for (int x = tid; x < 16 * HID; x += 256) S[x] = 0.f;
  __syncthreads();

  for (int it = 0; it < 24; ++it) {
    if (wave == 0) {
      // out1 tile: x=e(32, contig), y=il (stride 32), z=jl (stride 384*32)
      unsigned long long base1 = (unsigned long long)(uintptr_t)out1 +
          2ull * (((size_t)(jb * 16) * NREL + (size_t)it * 16) * 32);
      // out2 tile: x=e(32, contig), y=il (stride 384*32), z=jl (stride 32)
      unsigned long long base2 = (unsigned long long)(uintptr_t)out2 +
          2ull * (((size_t)(it * 16) * NREL + (size_t)jb * 16) * 32);
      unsigned int atB = (unsigned int)(uintptr_t)At;
      tdm_load_tile_f16(atB,       base1, 32u,          (unsigned)(NREL * 32));
      tdm_load_tile_f16(atB + 64u, base2, (unsigned)(NREL * 32), 32u);
      __builtin_amdgcn_s_wait_tensorcnt(0);
    }
    __syncthreads();
    for (int mt = wave * 2; mt < wave * 2 + 2; ++mt) {
      half16 a0 = load_a_frag(At + mt * 16 * 64, 64);
      half16 a1 = load_a_frag(At + mt * 16 * 64 + 32, 64);
      for (int nt = 0; nt < 32; ++nt) {
        half16 b0 = load_b_frag_cm(BwT + nt * 16 * 64, 64);
        half16 b1 = load_b_frag_cm(BwT + nt * 16 * 64 + 32, 64);
        floatx8 acc = {};
        acc = wmma_f16(a0, b0, acc);
        acc = wmma_f16(a1, b1, acc);
        int c = nt * 16 + (lane & 15);
        float bias = bv[c];
        float l1 = 0.f, l2 = 0.f, ssum = 0.f;
#pragma unroll
        for (int r = 0; r < 8; ++r) {
          float ax = fmaxf(acc[r] + bias, 0.f);
          l1 += ax; l2 += ax * ax; ssum += ax;
        }
        // all 8 rows of this lane share jl == mt (p = jl*16 + il)
        atomicAdd(&S[mt * HID + c], ssum);
        atomicAdd(&s1[c], l1);
        atomicAdd(&s2[c], l2);
      }
    }
    __syncthreads();
  }
  for (int x = tid; x < 16 * HID; x += 256) {
    int jl = x >> 9, c = x & (HID - 1);
    Sglob[(size_t)(jb * 16 + jl) * HID + c] = S[x];
  }
  for (int c = tid; c < HID; c += 256) {
    atomicAdd(&chanSum[c], s1[c]);
    atomicAdd(&chanSumSq[c], s2[c]);
  }
}

// pcT[j,c] = (S[j,c] - 384*m_c)*rsqrt(v_c+eps)*g[c] + 384*be[c]
__global__ void pp_finalize_pc_kernel(const float* __restrict__ Sglob,
                                      const float* __restrict__ chanSum,
                                      const float* __restrict__ chanSumSq,
                                      const float* __restrict__ g,
                                      const float* __restrict__ be,
                                      float* __restrict__ pcT) {
  int idx = blockIdx.x * 256 + threadIdx.x;
  if (idx >= NREL * HID) return;
  int c = idx & (HID - 1);
  const float inv = 1.f / ((float)NREL * (float)NREL);
  float m = chanSum[c] * inv;
  float var = chanSumSq[c] * inv - m * m;
  pcT[idx] = (Sglob[idx] - (float)NREL * m) * rsqrtf(var + 1e-5f) * g[c] +
             (float)NREL * be[c];
}

__global__ void pp_visual_kernel(float* __restrict__ gate,
                                 const float* __restrict__ uni) {
  int idx = blockIdx.x * 256 + threadIdx.x;   // f32x4 chunks
  if (idx >= NREL * POOL / 4) return;
  f32x4 g = *(const f32x4*)(gate + (size_t)idx * 4);
  f32x4 u = *(const f32x4*)(uni + (size_t)idx * 4);
  *(f32x4*)(gate + (size_t)idx * 4) = g * u;
}

__global__ void pp_onehot_kernel(const int* __restrict__ labels, float* __restrict__ out) {
  int idx = blockIdx.x * 256 + threadIdx.x;
  if (idx >= NOBJ * NOBJCLS) return;
  int r = idx / NOBJCLS, c = idx % NOBJCLS;
  out[idx] = (labels[r] == c) ? 1.f : 0.f;
}

__global__ void pp_zero_kernel(float* __restrict__ p, int n) {
  int idx = blockIdx.x * 256 + threadIdx.x;
  if (idx < n) p[idx] = 0.f;
}

// ---------------------------------------------------------------------------
// Launcher
// ---------------------------------------------------------------------------
extern "C" void kernel_launch(void* const* d_in, const int* in_sizes, int n_in,
                              void* d_out, int out_size, void* d_ws, size_t ws_size,
                              hipStream_t stream) {
  const float* roi   = (const float*)d_in[0];
  const float* uni   = (const float*)d_in[1];
  const int*   pairs = (const int*)d_in[2];
  const int*   labs  = (const int*)d_in[3];
  const float* W_obj = (const float*)d_in[4];
  const float* b_obj = (const float*)d_in[5];
  const float* g_obj = (const float*)d_in[6];
  const float* be_obj= (const float*)d_in[7];
  const float* W_a   = (const float*)d_in[8];
  const float* b_a   = (const float*)d_in[9];
  const float* g_a   = (const float*)d_in[10];
  const float* be_a  = (const float*)d_in[11];
  const float* W_b   = (const float*)d_in[12];
  const float* b_b   = (const float*)d_in[13];
  const float* g_b   = (const float*)d_in[14];
  const float* be_b  = (const float*)d_in[15];
  const float* g_pair= (const float*)d_in[16];
  const float* be_pair=(const float*)d_in[17];
  const float* Wq1   = (const float*)d_in[18];
  const float* Wkv1  = (const float*)d_in[19];
  const float* Wo1   = (const float*)d_in[20];
  const float* bo1   = (const float*)d_in[21];
  const float* Wq2   = (const float*)d_in[22];
  const float* Wkv2  = (const float*)d_in[23];
  const float* Wo2   = (const float*)d_in[24];
  const float* bo2   = (const float*)d_in[25];
  const float* W_post= (const float*)d_in[26];
  const float* b_post= (const float*)d_in[27];
  const float* W_rel = (const float*)d_in[28];
  const float* b_rel = (const float*)d_in[29];
  const float* W_ctx = (const float*)d_in[30];
  const float* b_ctx = (const float*)d_in[31];

  char* ws = (char*)d_ws;
  size_t off = 0;
  auto alloc = [&](size_t bytes) { char* p = ws + off; off += (bytes + 255) & ~(size_t)255; return p; };
  float*    objpre   = (float*)alloc(NOBJ * HID * 4);
  float*    obj_ctx  = (float*)alloc(NOBJ * HID * 4);
  float*    concat_so= (float*)alloc((size_t)NREL * 1024 * 4);
  _Float16* sh       = (_Float16*)alloc((size_t)NREL * HID * 2);
  _Float16* oh       = (_Float16*)alloc((size_t)NREL * HID * 2);
  _Float16* out1     = (_Float16*)alloc((size_t)NREL * NREL * 32 * 2);
  _Float16* out2     = (_Float16*)alloc((size_t)NREL * NREL * 32 * 2);
  float*    Sglob    = (float*)alloc((size_t)NREL * HID * 4);
  float*    chanSum  = (float*)alloc(HID * 4);
  float*    chanSumSq= (float*)alloc(HID * 4);
  float*    pcT      = (float*)alloc((size_t)NREL * HID * 4);
  float*    fs_pre   = (float*)alloc((size_t)NREL * HID * 4);
  float*    fo_pre   = (float*)alloc((size_t)NREL * HID * 4);
  float*    ctx_rois = (float*)alloc((size_t)NREL * HID * 4);
  float*    gate     = (float*)alloc((size_t)NREL * POOL * 4);
  (void)ws_size; (void)n_in; (void)in_sizes; (void)out_size;

  float* rel_out = (float*)d_out;                       // [384][51]
  float* obj_out = (float*)d_out + NREL * NRELCLS;      // [128][151]

  const size_t attn_lds = (size_t)(NREL*32 + 96*32 + NREL*32 + NREL*32 + 32*NREL + 8*16*32) * 2;
  const size_t comb_lds = (size_t)(64*HID + 256*64) * 2 + (size_t)(16*HID + 3*HID) * 4;

  // 1) obj_ctx = bn1d(relu(roi @ W_obj + b_obj))
  pp_gemm_kernel<<<dim3(HID/32, NOBJ/32), 128, 0, stream>>>(
      roi, POOL, W_obj, HID, b_obj, objpre, HID, NOBJ, HID, POOL, 1, 0);
  pp_bn1d_kernel<<<dim3((HID+255)/256), 256, 0, stream>>>(objpre, obj_ctx, NOBJ, HID, g_obj, be_obj);

  // 2) gather s, o
  pp_gather_kernel<<<dim3(NREL*HID/8/256), 256, 0, stream>>>(obj_ctx, pairs, concat_so, sh, oh);

  // 3) axial attention, both branches (pre-Wo outputs)
  pp_axial_attn_kernel<<<dim3(NREL), 256, attn_lds, stream>>>(sh, oh, Wq1, Wkv1, out1);
  pp_axial_attn_kernel<<<dim3(NREL), 256, attn_lds, stream>>>(oh, sh, Wq2, Wkv2, out2);

  // 4) combine (TDM-staged) + bn2d stats + column-sum reduction
  pp_zero_kernel<<<dim3(4), 256, 0, stream>>>(chanSum, 2 * HID);  // chanSum|chanSumSq adjacent
  pp_combine_kernel<<<dim3(NREL/16), 256, comb_lds, stream>>>(
      out1, out2, Wo1, Wo2, bo1, bo2, Sglob, chanSum, chanSumSq);
  pp_finalize_pc_kernel<<<dim3(NREL*HID/256), 256, 0, stream>>>(
      Sglob, chanSum, chanSumSq, g_pair, be_pair, pcT);

  // 5) fs, fo, ctx_pairwise_rois = bn(fs)+bn(fo)+pcT
  pp_gemm_kernel<<<dim3(HID/32, NREL/32), 128, 0, stream>>>(
      concat_so, 1024, W_a, HID, b_a, fs_pre, HID, NREL, HID, HID, 1, 0);
  pp_gemm_kernel<<<dim3(HID/32, NREL/32), 128, 0, stream>>>(
      concat_so + HID, 1024, W_b, HID, b_b, fo_pre, HID, NREL, HID, HID, 1, 0);
  pp_bn_add_kernel<<<dim3(2), 256, 0, stream>>>(fs_pre, fo_pre, pcT, g_a, be_a, g_b, be_b, ctx_rois);

  // 6) gate = [s|o] @ W_post + b_post ; visual = union * gate (in place)
  pp_gemm_kernel<<<dim3(POOL/32, NREL/32), 128, 0, stream>>>(
      concat_so, 1024, W_post, POOL, b_post, gate, POOL, NREL, POOL, 1024, 0, 0);
  pp_visual_kernel<<<dim3(NREL*POOL/4/256), 256, 0, stream>>>(gate, uni);

  // 7) rel_dists = visual @ W_rel + b_rel + ctx_rois @ W_ctx + b_ctx
  pp_gemm_kernel<<<dim3((NRELCLS+31)/32, NREL/32), 128, 0, stream>>>(
      gate, POOL, W_rel, NRELCLS, b_rel, rel_out, NRELCLS, NREL, NRELCLS, POOL, 0, 0);
  pp_gemm_kernel<<<dim3((NRELCLS+31)/32, NREL/32), 128, 0, stream>>>(
      ctx_rois, HID, W_ctx, NRELCLS, b_ctx, rel_out, NRELCLS, NREL, NRELCLS, HID, 0, 1);

  // 8) obj_dists = one_hot(labels)
  pp_onehot_kernel<<<dim3((NOBJ*NOBJCLS+255)/256), 256, 0, stream>>>(labs, obj_out);
}
